// MultiheadSelfAttention_81475529605301
// MI455X (gfx1250) — matrix operand order repaired and checked by reading, not compiled, not run on hardware
//
#include <hip/hip_runtime.h>

// ---------------------------------------------------------------------------
// MI455X (gfx1250) multi-head self-attention with RoPE, causal mask.
// bf16 operands + f32 accumulation via v_wmma_f32_16x16x32_bf16 (wave32),
// Tensor Data Mover (tensor_load_to_lds + s_wait_tensorcnt) for K/V tiles.
// ---------------------------------------------------------------------------

typedef __attribute__((ext_vector_type(16))) __bf16 v16bf;
typedef __attribute__((ext_vector_type(8)))  float  v8f;
typedef __attribute__((ext_vector_type(4)))  unsigned int u32x4;
typedef __attribute__((ext_vector_type(8)))  int          i32x8;
typedef __attribute__((ext_vector_type(4)))  int          i32x4;
typedef unsigned short ushort_t;

#define B_  4
#define S_  1024
#define D_  1024
#define H_  16
#define DK_ 64

union FragU { v16bf v; uint4 q[2]; };

// A-fragment (16x32, row-major source, row stride ld elements):
// lane<16: row=lane,  K = {0..7, 16..23}; lane>=16: row=lane-16, K = {8..15, 24..31}
__device__ __forceinline__ v16bf load_frag_a(const ushort_t* base, int ld, int lane) {
    const ushort_t* p = base + (size_t)(lane & 15) * ld + ((lane >> 4) << 3);
    FragU u;
    u.q[0] = *(const uint4*)(p);
    u.q[1] = *(const uint4*)(p + 16);
    return u.v;
}

// B-fragment (32x16, K-contiguous source: row n holds K values, row stride ld):
// lane<16: col=lane, K = {0..7, 16..23}; lane>=16: col=lane-16, K = {8..15, 24..31}
__device__ __forceinline__ v16bf load_frag_b(const ushort_t* base, int ld, int lane) {
    const ushort_t* p = base + (size_t)(lane & 15) * ld + ((lane >> 4) << 3);
    FragU u;
    u.q[0] = *(const uint4*)(p);
    u.q[1] = *(const uint4*)(p + 16);
    return u.v;
}

__device__ __forceinline__ v8f wmma_bf16(v16bf a, v16bf b, v8f c) {
    return __builtin_amdgcn_wmma_f32_16x16x32_bf16(false, a, false, b, (short)0, c,
                                                   false, false);
}

// ---------------------------------------------------------------------------
// Tensor Data Mover: 2D tile load (data_size = 8B units) per ISA cdna5 §8.
// g0: count=1 | lds_addr | global_addr(57b) | type=2
// g1: data_size=3(8B), tensor_dim0/1, tile_dim0/1, tensor_dim0_stride
// ---------------------------------------------------------------------------
__device__ __forceinline__ void tdm_load_2d(unsigned int lds_addr,
                                            unsigned long long gaddr,
                                            unsigned int td0, unsigned int td1,
                                            unsigned int tile0, unsigned int tile1,
                                            unsigned long long stride0) {
    u32x4 g0;
    g0[0] = 1u;                                             // count=1, user mode
    g0[1] = lds_addr;                                       // LDS byte address
    g0[2] = (unsigned int)(gaddr & 0xFFFFFFFFull);          // global_addr[31:0]
    g0[3] = (unsigned int)((gaddr >> 32) & 0x1FFFFFFull)    // global_addr[56:32]
            | (2u << 30);                                   // type = 2 ("image")
    i32x8 g1;
    g1[0] = (int)(3u << 16);                                // data_size = 8B
    g1[1] = (int)((td0 & 0xFFFFu) << 16);                   // tensor_dim0[15:0]
    g1[2] = (int)(((td0 >> 16) & 0xFFFFu) | ((td1 & 0xFFFFu) << 16));
    g1[3] = (int)(((td1 >> 16) & 0xFFFFu) | ((tile0 & 0xFFFFu) << 16));
    g1[4] = (int)(tile1 & 0xFFFFu);                         // tile_dim1 (tile_dim2=0)
    g1[5] = (int)(stride0 & 0xFFFFFFFFull);                 // dim0 stride lo
    g1[6] = (int)((stride0 >> 32) & 0xFFFFull);             // dim0 stride hi
    g1[7] = 0;
    i32x4 z4 = {0, 0, 0, 0};
#if defined(__clang_major__) && (__clang_major__ >= 23)
    i32x8 z8 = {0, 0, 0, 0, 0, 0, 0, 0};
    __builtin_amdgcn_tensor_load_to_lds(g0, g1, z4, z4, z8, 0);
#else
    __builtin_amdgcn_tensor_load_to_lds(g0, g1, z4, z4, 0);
#endif
}

// ---------------------------------------------------------------------------
// f32 -> bf16 convert
// ---------------------------------------------------------------------------
__global__ void cvt_f32_bf16(const float* __restrict__ src, __bf16* __restrict__ dst, int n) {
    for (int i = blockIdx.x * blockDim.x + threadIdx.x; i < n; i += gridDim.x * blockDim.x)
        dst[i] = (__bf16)src[i];
}

// ---------------------------------------------------------------------------
// GEMM: C[M,N] = A[M,K] (bf16, row-major) * W[N,K]^T (bf16, N-major / K-contig)
// MODE 0: f32 row-major out (final projection)
// MODE 1: bf16 out scattered to [B][H][S][DK]   (Q, K pre-RoPE)
// MODE 2: bf16 out scattered to [B][H][DK][S]   (V transposed)
// Block = 128 threads = 4 waves; wave tile 32x64, block tile 64x128.
// ---------------------------------------------------------------------------
template <int MODE>
__global__ __launch_bounds__(128)
void gemm_bf16(const ushort_t* __restrict__ A, const ushort_t* __restrict__ Bw,
               float* __restrict__ Cf, __bf16* __restrict__ Cb,
               int M, int N, int K) {
    const int lane  = threadIdx.x & 31;
    const int w     = threadIdx.x >> 5;
    const int tileM = blockIdx.y * 64  + (w >> 1) * 32;
    const int tileN = blockIdx.x * 128 + (w & 1) * 64;

    v8f acc[2][4] = {};
    for (int k0 = 0; k0 < K; k0 += 32) {
        v16bf a0 = load_frag_a(A + (size_t)tileM * K + k0, K, lane);
        v16bf a1 = load_frag_a(A + (size_t)(tileM + 16) * K + k0, K, lane);
        v16bf b[4];
#pragma unroll
        for (int j = 0; j < 4; ++j)
            b[j] = load_frag_b(Bw + (size_t)(tileN + 16 * j) * K + k0, K, lane);
#pragma unroll
        for (int j = 0; j < 4; ++j) {
            acc[0][j] = wmma_bf16(a0, b[j], acc[0][j]);
            acc[1][j] = wmma_bf16(a1, b[j], acc[1][j]);
        }
    }

    const int hi   = lane >> 4;
    const int col0 = lane & 15;
#pragma unroll
    for (int i = 0; i < 2; ++i)
#pragma unroll
        for (int j = 0; j < 4; ++j)
#pragma unroll
            for (int r = 0; r < 8; ++r) {
                int row = tileM + 16 * i + r + 8 * hi;
                int col = tileN + 16 * j + col0;
                float v = acc[i][j][r];
                if (MODE == 0) {
                    Cf[(size_t)row * N + col] = v;
                } else {
                    int b_ = row >> 10, s = row & (S_ - 1);
                    int h  = col >> 6,  d = col & (DK_ - 1);
                    if (MODE == 1)
                        Cb[(((size_t)b_ * H_ + h) * S_ + s) * DK_ + d] = (__bf16)v;
                    else
                        Cb[(((size_t)b_ * H_ + h) * DK_ + d) * S_ + s] = (__bf16)v;
                }
            }
}

// ---------------------------------------------------------------------------
// In-place RoPE on bf16 tensor laid out [B][H][S][DK]; one thread per pair.
// ---------------------------------------------------------------------------
__global__ void rope_kernel(__bf16* __restrict__ t, const int* __restrict__ pos, int npairs) {
    int idx = blockIdx.x * blockDim.x + threadIdx.x;
    if (idx >= npairs) return;
    int i  = idx & 31;
    int s  = (idx >> 5) & (S_ - 1);
    int bh = idx >> 15;
    float p    = (float)pos[s];
    float freq = __expf(-(float)i * 0.28782313662425574f);   // theta^(-2i/DK)
    float ang  = p * freq;
    float sn, cs;
    sincosf(ang, &sn, &cs);
    __bf16* q = t + (((size_t)bh * S_) + s) * DK_ + 2 * i;
    float e = (float)q[0], o = (float)q[1];
    q[0] = (__bf16)(e * cs - o * sn);
    q[1] = (__bf16)(e * sn + o * cs);
}

// ---------------------------------------------------------------------------
// Flash attention (causal). Grid = (S/64, B*H), block = 128 threads (4 waves).
// K (32x64) and V (64x32) tiles staged once per block in LDS by the TDM
// (issued by wave 0, published with s_wait_tensorcnt + barrier); all 4 waves
// consume them via ds_load fragments. P round-trips through per-wave LDS to
// convert C-layout -> A-layout for the P·V WMMAs.
// ---------------------------------------------------------------------------
__global__ __launch_bounds__(128)
void attn_kernel(const ushort_t* __restrict__ Q, const ushort_t* __restrict__ Kt,
                 const ushort_t* __restrict__ Vt, __bf16* __restrict__ O) {
    __shared__ __align__(16) ushort_t Kbuf[32 * 64];       // [krow][d]   4 KB
    __shared__ __align__(16) ushort_t Vbuf[64 * 32];       // [d][krow]   4 KB
    __shared__ __align__(16) ushort_t Pbuf[4 * 16 * 32];   // per-wave P  4 KB

    const int lane  = threadIdx.x & 31;
    const int w     = threadIdx.x >> 5;
    const int bh    = blockIdx.y;
    const int qbase = blockIdx.x * 64 + w * 16;
    const int hi    = lane >> 4;
    const int nlo   = lane & 15;

    const ushort_t* Qp = Q  + (size_t)bh * S_ * DK_;
    const ushort_t* Kp = Kt + (size_t)bh * S_ * DK_;
    const ushort_t* Vp = Vt + (size_t)bh * DK_ * S_;
    ushort_t* myP = Pbuf + w * (16 * 32);

    // ISA §10.2: low 32 bits of a flat LDS address are the LDS byte offset.
    const unsigned int kbuf_lds = (unsigned int)(size_t)(void*)Kbuf;
    const unsigned int vbuf_lds = (unsigned int)(size_t)(void*)Vbuf;
    const unsigned long long kga = (unsigned long long)(size_t)Kp;
    const unsigned long long vga = (unsigned long long)(size_t)Vp;

    // Q fragments (DK=64 -> two K=32 chunks), kept in registers.
    v16bf qf0 = load_frag_a(Qp + (size_t)qbase * DK_,      DK_, lane);
    v16bf qf1 = load_frag_a(Qp + (size_t)qbase * DK_ + 32, DK_, lane);

    v8f o[4] = {};
    float mrow[8], lrow[8];
#pragma unroll
    for (int r = 0; r < 8; ++r) { mrow[r] = -1e30f; lrow[r] = 0.f; }

    const int kend = blockIdx.x * 64 + 63;   // same trip count for all 4 waves
    for (int kn0 = 0; kn0 <= kend; kn0 += 32) {
        // ---- TDM: stage K tile (contiguous 4KB) and V tile (64 rows x 64B) ----
        if (w == 0) {
            // K rows kn0..kn0+31, 64 bf16 each: 1-D copy of 512 x 8B units.
            tdm_load_2d(kbuf_lds, kga + (unsigned long long)kn0 * DK_ * 2,
                        1u << 20, 1u, 512u, 1u, 1u << 20);
            // V^T tile: 64 rows (d), 8 x 8B units each, row stride 256 units.
            tdm_load_2d(vbuf_lds, vga + (unsigned long long)kn0 * 2,
                        256u, 64u, 8u, 64u, 256u);
            __builtin_amdgcn_s_wait_tensorcnt(0);
        }
        __syncthreads();   // tiles visible to all waves

        // ---- scores: S[16q x 32k] = Q (16x64) . K^T, K from LDS ----
        v8f sc[2];
#pragma unroll
        for (int t = 0; t < 2; ++t) {
            v16bf kb0 = load_frag_b(Kbuf + (16 * t) * DK_,      DK_, lane);
            v16bf kb1 = load_frag_b(Kbuf + (16 * t) * DK_ + 32, DK_, lane);
            v8f z = {};
            z     = wmma_bf16(qf0, kb0, z);
            sc[t] = wmma_bf16(qf1, kb1, z);
        }

        // ---- scale, causal mask, online softmax (row reductions over 16 lanes)
        float p[2][8];
#pragma unroll
        for (int r = 0; r < 8; ++r) {
            int   qidx = qbase + r + 8 * hi;
            float s0 = sc[0][r] * 0.125f;                 // 1/sqrt(64)
            float s1 = sc[1][r] * 0.125f;
            if (kn0 + nlo      > qidx) s0 = -1e30f;
            if (kn0 + 16 + nlo > qidx) s1 = -1e30f;
            float rm = fmaxf(s0, s1);
            rm = fmaxf(rm, __shfl_xor(rm, 1));
            rm = fmaxf(rm, __shfl_xor(rm, 2));
            rm = fmaxf(rm, __shfl_xor(rm, 4));
            rm = fmaxf(rm, __shfl_xor(rm, 8));
            float mnew  = fmaxf(mrow[r], rm);
            float alpha = __expf(mrow[r] - mnew);
            float p0 = __expf(s0 - mnew);
            float p1 = __expf(s1 - mnew);
            float rs = p0 + p1;
            rs += __shfl_xor(rs, 1);
            rs += __shfl_xor(rs, 2);
            rs += __shfl_xor(rs, 4);
            rs += __shfl_xor(rs, 8);
            lrow[r] = lrow[r] * alpha + rs;
            mrow[r] = mnew;
            p[0][r] = p0;
            p[1][r] = p1;
#pragma unroll
            for (int t4 = 0; t4 < 4; ++t4) o[t4][r] *= alpha;
        }

        // ---- C-layout -> A-layout via per-wave LDS buffer (bf16) ----
#pragma unroll
        for (int t = 0; t < 2; ++t)
#pragma unroll
            for (int r = 0; r < 8; ++r) {
                union { __bf16 b; ushort_t u; } cv;
                cv.b = (__bf16)p[t][r];
                myP[(r + 8 * hi) * 32 + 16 * t + nlo] = cv.u;
            }
        __syncthreads();
        v16bf pf = load_frag_a(myP, 32, lane);

        // ---- O += P (16x32) . V (32x64); V fragments from LDS ----
#pragma unroll
        for (int t = 0; t < 4; ++t) {
            v16bf vf = load_frag_b(Vbuf + (16 * t) * 32, 32, lane);
            o[t] = wmma_bf16(pf, vf, o[t]);
        }
        __syncthreads();   // done with Kbuf/Vbuf/Pbuf before next TDM
    }

    // ---- normalize and store bf16 to [B][S][D] ----
    const int b = bh >> 4, h = bh & 15;
#pragma unroll
    for (int r = 0; r < 8; ++r) {
        float inv = 1.0f / lrow[r];
        int s = qbase + r + 8 * hi;
#pragma unroll
        for (int t = 0; t < 4; ++t)
            O[((size_t)b * S_ + s) * D_ + h * DK_ + 16 * t + nlo] =
                (__bf16)(o[t][r] * inv);
    }
}

// ---------------------------------------------------------------------------
// Host launcher
// ---------------------------------------------------------------------------
extern "C" void kernel_launch(void* const* d_in, const int* in_sizes, int n_in,
                              void* d_out, int out_size, void* d_ws, size_t ws_size,
                              hipStream_t stream) {
    (void)in_sizes; (void)n_in; (void)out_size; (void)ws_size;
    const float* x  = (const float*)d_in[0];
    const float* wq = (const float*)d_in[1];
    const float* wk = (const float*)d_in[2];
    const float* wv = (const float*)d_in[3];
    const float* wo = (const float*)d_in[4];
    const int*   tp = (const int*)d_in[5];
    float* out = (float*)d_out;

    char* ws = (char*)d_ws;
    __bf16* xb  = (__bf16*)(ws + (size_t)0);
    __bf16* wqb = (__bf16*)(ws + ((size_t)8  << 20));
    __bf16* wkb = (__bf16*)(ws + ((size_t)10 << 20));
    __bf16* wvb = (__bf16*)(ws + ((size_t)12 << 20));
    __bf16* wob = (__bf16*)(ws + ((size_t)14 << 20));
    __bf16* Qr  = (__bf16*)(ws + ((size_t)16 << 20));  // [B][H][S][DK]
    __bf16* Kr  = (__bf16*)(ws + ((size_t)24 << 20));  // [B][H][S][DK]
    __bf16* Vt  = (__bf16*)(ws + ((size_t)32 << 20));  // [B][H][DK][S]
    __bf16* Ob  = (__bf16*)(ws + ((size_t)40 << 20));  // [B][S][D]

    const int NX = B_ * S_ * D_;   // 4,194,304
    const int NW = D_ * D_;        // 1,048,576
    cvt_f32_bf16<<<2048, 256, 0, stream>>>(x,  xb,  NX);
    cvt_f32_bf16<<<1024, 256, 0, stream>>>(wq, wqb, NW);
    cvt_f32_bf16<<<1024, 256, 0, stream>>>(wk, wkb, NW);
    cvt_f32_bf16<<<1024, 256, 0, stream>>>(wv, wvb, NW);
    cvt_f32_bf16<<<1024, 256, 0, stream>>>(wo, wob, NW);

    dim3 gg(D_ / 128, (B_ * S_) / 64);   // (8, 64)
    gemm_bf16<1><<<gg, 128, 0, stream>>>((const ushort_t*)xb, (const ushort_t*)wqb,
                                         nullptr, Qr, B_ * S_, D_, D_);
    gemm_bf16<1><<<gg, 128, 0, stream>>>((const ushort_t*)xb, (const ushort_t*)wkb,
                                         nullptr, Kr, B_ * S_, D_, D_);
    gemm_bf16<2><<<gg, 128, 0, stream>>>((const ushort_t*)xb, (const ushort_t*)wvb,
                                         nullptr, Vt, B_ * S_, D_, D_);

    const int npairs = B_ * H_ * S_ * (DK_ / 2);   // 2,097,152
    rope_kernel<<<npairs / 256, 256, 0, stream>>>(Qr, tp, npairs);
    rope_kernel<<<npairs / 256, 256, 0, stream>>>(Kr, tp, npairs);

    dim3 ga(S_ / 64, B_ * H_);           // (16, 64)
    attn_kernel<<<ga, 128, 0, stream>>>((const ushort_t*)Qr, (const ushort_t*)Kr,
                                        (const ushort_t*)Vt, Ob);

    gemm_bf16<0><<<gg, 128, 0, stream>>>((const ushort_t*)Ob, (const ushort_t*)wob,
                                         out, nullptr, B_ * S_, D_, D_);
}
